// PointNetSetAbstractionWithOriginalGraph_76398878261706
// MI455X (gfx1250) — matrix operand order repaired and checked by reading, not compiled, not run on hardware
//
#include <hip/hip_runtime.h>
#include <hip/hip_bf16.h>

// ---------------------------------------------------------------------------
// DGCNN-style PointNet set abstraction for MI455X (gfx1250, wave32, WMMA).
// All GEMMs run on v_wmma_f32_16x16x32_f16 (f16 in, f32 accumulate).
// Each wave computes a 16x64 output strip (4 accumulators) so one A fragment
// feeds 4 WMMAs per K-chunk -> 4x less A traffic / gather VALU than 1 tile.
// BN biases cancel inside batch-norm ((x+b)-mean(x+b) == x-mean(x)), so
// bias0/1/2 are mathematically dead and not read.
// ---------------------------------------------------------------------------

typedef __attribute__((ext_vector_type(16))) _Float16 v16h;
typedef __attribute__((ext_vector_type(8)))  _Float16 v8h;
typedef __attribute__((ext_vector_type(8)))  float    v8f;

union FragA { v16h v; v8h h[2]; };

#define NB    4
#define NPTS  4096
#define DF    64
#define KNN   24
#define GOUT  128
#define NKR   (NPTS * KNN)   // 98304 edge rows per batch
#define BNR   (NB * NPTS)    // 16384 point rows total
#define EPSV  1e-5f

// ---------------------------------------------------------------------------
// Prep: transpose/convert activations + weights to f16 layouts WMMA can eat.
// ---------------------------------------------------------------------------
__global__ __launch_bounds__(256) void prep_pts_kernel(
    const float* __restrict__ pts, _Float16* __restrict__ ptsH) {
  size_t t = (size_t)blockIdx.x * blockDim.x + threadIdx.x;
  if (t >= (size_t)NB * NPTS * DF) return;
  int c = (int)(t & (DF - 1));
  size_t bn = t >> 6;
  int n = (int)(bn & (NPTS - 1));
  int b = (int)(bn >> 12);
  ptsH[t] = (_Float16)pts[((size_t)b * DF + c) * NPTS + n];
}

__global__ __launch_bounds__(256) void prep_g0_kernel(
    const float* __restrict__ xyz, const float* __restrict__ pts,
    _Float16* __restrict__ g0H) {
  size_t t = (size_t)blockIdx.x * blockDim.x + threadIdx.x;
  if (t >= (size_t)BNR * 96) return;
  int c = (int)(t % 96);
  size_t bn = t / 96;
  int n = (int)(bn & (NPTS - 1));
  int b = (int)(bn >> 12);
  float v = 0.0f;
  if (c < 3)       v = xyz[((size_t)b * 3 + c) * NPTS + n];
  else if (c < 67) v = pts[((size_t)b * DF + (c - 3)) * NPTS + n];
  g0H[t] = (_Float16)v;
}

__global__ __launch_bounds__(256) void prep_w_kernel(
    const float* __restrict__ w, _Float16* __restrict__ wh,
    int O, int Cin, int Cpad) {
  int t = blockIdx.x * blockDim.x + threadIdx.x;
  if (t >= O * Cpad) return;
  int c = t % Cpad, o = t / Cpad;
  wh[t] = (_Float16)((c < Cin) ? w[(size_t)o * Cin + c] : 0.0f);
}

// ---------------------------------------------------------------------------
// kNN: one thread per query point, candidates tiled through LDS,
// insertion-sorted top-(K+1); drop entry 0 (self, exact distance 0).
// ---------------------------------------------------------------------------
#define KNN_TILE 256
__global__ __launch_bounds__(128) void knn_kernel(
    const float* __restrict__ xyz, int* __restrict__ edges) {
  int b = blockIdx.y;
  int n = blockIdx.x * blockDim.x + threadIdx.x;
  const float* xb = xyz + (size_t)b * 3 * NPTS;
  float qx = xb[n], qy = xb[NPTS + n], qz = xb[2 * NPTS + n];

  float best[KNN + 1];
  int   bidx[KNN + 1];
  for (int i = 0; i <= KNN; ++i) { best[i] = __builtin_inff(); bidx[i] = 0; }

  __shared__ float sx[KNN_TILE], sy[KNN_TILE], sz[KNN_TILE];
  for (int t0 = 0; t0 < NPTS; t0 += KNN_TILE) {
    __syncthreads();
    for (int i = threadIdx.x; i < KNN_TILE; i += blockDim.x) {
      sx[i] = xb[t0 + i];
      sy[i] = xb[NPTS + t0 + i];
      sz[i] = xb[2 * NPTS + t0 + i];
    }
    __syncthreads();
    for (int i = 0; i < KNN_TILE; ++i) {
      float dx = qx - sx[i], dy = qy - sy[i], dz = qz - sz[i];
      float d = dx * dx + dy * dy + dz * dz;
      if (d < best[KNN]) {
        int j = KNN;
        while (j > 0 && best[j - 1] > d) {
          best[j] = best[j - 1]; bidx[j] = bidx[j - 1]; --j;
        }
        best[j] = d; bidx[j] = t0 + i;
      }
    }
  }
  int* ep = edges + ((size_t)b * NPTS + n) * KNN;
  for (int k = 0; k < KNN; ++k) ep[k] = bidx[k + 1];
}

// ---------------------------------------------------------------------------
// WMMA fragment layouts used here (16x16x32 f16, wave32; A/B k order is this code's choice):
//   A: lane row = lane&15; elem i    -> K = 16g + i   (two v8h halves)
//   B: lane col = lane&15; elem i    -> K = 16g + i   (one contiguous v16h)
//   C: VGPR v -> M = v + 8g, N = lane&15
// ---------------------------------------------------------------------------

#define WMMA_F16(Av, Bv, Cv) \
  __builtin_amdgcn_wmma_f32_16x16x32_f16(false, (Av), false, (Bv), (short)0, (Cv), false, false)

// EdgeConv: rows are (n,k) pairs; feature c<64 = center, c>=64 = nbr - center.
// One wave -> 16 rows x 64 output channels (4 accumulators).
__global__ __launch_bounds__(32) void edge_wmma_kernel(
    const _Float16* __restrict__ ptsH,  // (B, N, 64) f16
    const _Float16* __restrict__ Wh,    // (128, 128) f16 row-major
    const int* __restrict__ edges,      // (B, N, K)
    _Float16* __restrict__ H) {         // (B, N*K, 128) f16
  const int lane = threadIdx.x & 31;
  const int g = lane >> 4, r = lane & 15;
  const int b = blockIdx.z;
  const int rowg = blockIdx.x * 16 + r;          // < 98304
  const int n = rowg / KNN;
  const int j = edges[(size_t)b * NKR + rowg];
  const _Float16* ctr = ptsH + ((size_t)b * NPTS + n) * DF;
  const _Float16* nbr = ptsH + ((size_t)b * NPTS + j) * DF;
  const int ocol0 = blockIdx.y * 64 + r;         // first of 4 col tiles
  const _Float16* wrow = Wh + (size_t)ocol0 * 128;

  v8f acc0 = {}, acc1 = {}, acc2 = {}, acc3 = {};
#pragma unroll
  for (int kc = 0; kc < 4; ++kc) {
    FragA a;
    const int c0 = kc * 32 + 16 * g;
    const int c1 = c0 + 8;
    if (c0 < DF) a.h[0] = *(const v8h*)(ctr + c0);
    else         a.h[0] = *(const v8h*)(nbr + (c0 - DF)) - *(const v8h*)(ctr + (c0 - DF));
    if (c1 < DF) a.h[1] = *(const v8h*)(ctr + c1);
    else         a.h[1] = *(const v8h*)(nbr + (c1 - DF)) - *(const v8h*)(ctr + (c1 - DF));
    const int wo = kc * 32 + 16 * g;
    v16h b0 = *(const v16h*)(wrow + wo);
    v16h b1 = *(const v16h*)(wrow + 16 * 128 + wo);
    v16h b2 = *(const v16h*)(wrow + 32 * 128 + wo);
    v16h b3 = *(const v16h*)(wrow + 48 * 128 + wo);
    acc0 = WMMA_F16(a.v, b0, acc0);
    acc1 = WMMA_F16(a.v, b1, acc1);
    acc2 = WMMA_F16(a.v, b2, acc2);
    acc3 = WMMA_F16(a.v, b3, acc3);
  }
  const size_t rbase = (size_t)blockIdx.x * 16 + 8 * g;
  _Float16* hp = H + ((size_t)b * NKR + rbase) * 128 + ocol0;
#pragma unroll
  for (int v = 0; v < 8; ++v) {
    hp[(size_t)v * 128 +  0] = (_Float16)acc0[v];
    hp[(size_t)v * 128 + 16] = (_Float16)acc1[v];
    hp[(size_t)v * 128 + 32] = (_Float16)acc2[v];
    hp[(size_t)v * 128 + 48] = (_Float16)acc3[v];
  }
}

// Generic Z = A(M x 32*kChunks) @ W^T, f16 in, f16 z out (stats come later).
// One wave -> 16 rows x 64 output channels.
__global__ __launch_bounds__(32) void gemm_wmma_kernel(
    const _Float16* __restrict__ A, int lda,
    const _Float16* __restrict__ W, int ldw,
    _Float16* __restrict__ Z, int ldz, int kChunks) {
  const int lane = threadIdx.x & 31;
  const int g = lane >> 4, r = lane & 15;
  const size_t row = (size_t)blockIdx.x * 16 + r;
  const int ocol0 = blockIdx.y * 64 + r;
  const _Float16* arow = A + row * lda;
  const _Float16* wrow = W + (size_t)ocol0 * ldw;

  v8f acc0 = {}, acc1 = {}, acc2 = {}, acc3 = {};
  for (int kc = 0; kc < kChunks; ++kc) {
    FragA a;
    const int c0 = kc * 32 + 16 * g;
    a.h[0] = *(const v8h*)(arow + c0);
    a.h[1] = *(const v8h*)(arow + c0 + 8);
    const int wo = kc * 32 + 16 * g;
    v16h b0 = *(const v16h*)(wrow + wo);
    v16h b1 = *(const v16h*)(wrow + (size_t)16 * ldw + wo);
    v16h b2 = *(const v16h*)(wrow + (size_t)32 * ldw + wo);
    v16h b3 = *(const v16h*)(wrow + (size_t)48 * ldw + wo);
    acc0 = WMMA_F16(a.v, b0, acc0);
    acc1 = WMMA_F16(a.v, b1, acc1);
    acc2 = WMMA_F16(a.v, b2, acc2);
    acc3 = WMMA_F16(a.v, b3, acc3);
  }
  const size_t rbase = (size_t)blockIdx.x * 16 + 8 * g;
  _Float16* zp = Z + rbase * ldz + ocol0;
#pragma unroll
  for (int v = 0; v < 8; ++v) {
    zp[(size_t)v * ldz +  0] = (_Float16)acc0[v];
    zp[(size_t)v * ldz + 16] = (_Float16)acc1[v];
    zp[(size_t)v * ldz + 32] = (_Float16)acc2[v];
    zp[(size_t)v * ldz + 48] = (_Float16)acc3[v];
  }
}

// ---------------------------------------------------------------------------
// Deterministic per-channel BN stats: block per channel, fixed-tree reduce.
// Emits fused affine: scale = gamma*rsqrt(var+eps), shift = beta - mean*scale.
// ---------------------------------------------------------------------------
__global__ __launch_bounds__(256) void col_stats_kernel(
    const _Float16* __restrict__ Z, int C, size_t M,
    const float* __restrict__ gamma, const float* __restrict__ beta,
    float* __restrict__ scale, float* __restrict__ shift) {
  const int c = blockIdx.x;
  __shared__ float ssum[256], ssq[256];
  float s = 0.0f, q = 0.0f;
  for (size_t m = threadIdx.x; m < M; m += 256) {
    float v = (float)Z[m * C + c];
    s += v; q += v * v;
  }
  ssum[threadIdx.x] = s; ssq[threadIdx.x] = q;
  __syncthreads();
  for (int off = 128; off > 0; off >>= 1) {
    if ((int)threadIdx.x < off) {
      ssum[threadIdx.x] += ssum[threadIdx.x + off];
      ssq[threadIdx.x]  += ssq[threadIdx.x + off];
    }
    __syncthreads();
  }
  if (threadIdx.x == 0) {
    float inv = 1.0f / (float)M;
    float mean = ssum[0] * inv;
    float var = ssq[0] * inv - mean * mean;
    float sc = gamma[c] * rsqrtf(var + EPSV);
    scale[c] = sc;
    shift[c] = beta[c] - mean * sc;
  }
}

// EdgeConv epilogue: affine + leaky, then max over the K neighbors.
__global__ __launch_bounds__(256) void ec_apply_max_kernel(
    const _Float16* __restrict__ H, const float* __restrict__ scale,
    const float* __restrict__ shift, _Float16* __restrict__ gfeat) {
  size_t t = (size_t)blockIdx.x * blockDim.x + threadIdx.x;
  if (t >= (size_t)BNR * GOUT) return;
  int o = (int)(t & (GOUT - 1));
  size_t bn = t >> 7;
  const _Float16* hp = H + bn * KNN * GOUT + o;
  float sc = scale[o], sh = shift[o];
  float m = -__builtin_inff();
  for (int k = 0; k < KNN; ++k) {
    float v = (float)hp[(size_t)k * GOUT] * sc + sh;
    v = (v > 0.0f) ? v : 0.2f * v;
    m = fmaxf(m, v);
  }
  gfeat[t] = (_Float16)m;
}

// Elementwise affine + (leaky)relu for gm / MLP layers.  leak=0 -> relu.
__global__ __launch_bounds__(256) void apply_act_kernel(
    const _Float16* __restrict__ Z, const float* __restrict__ scale,
    const float* __restrict__ shift, _Float16* __restrict__ A,
    size_t total, int cmask, float leak) {
  size_t t = (size_t)blockIdx.x * blockDim.x + threadIdx.x;
  if (t >= total) return;
  int o = (int)(t & (size_t)cmask);
  float v = (float)Z[t] * scale[o] + shift[o];
  A[t] = (_Float16)((v > 0.0f) ? v : leak * v);
}

// Column max over the N points of each batch.
__global__ __launch_bounds__(256) void colmax_kernel(
    const _Float16* __restrict__ A, int C, float* __restrict__ out) {
  int t = blockIdx.x * blockDim.x + threadIdx.x;
  if (t >= NB * C) return;
  int o = t % C, b = t / C;
  const _Float16* p = A + (size_t)b * NPTS * C + o;
  float m = -__builtin_inff();
  for (int n = 0; n < NPTS; ++n) m = fmaxf(m, (float)p[(size_t)n * C]);
  out[t] = m;
}

// Final fusion: (B,384) @ fu_w^T -> BN over batch (4 samples) -> leaky.
__global__ __launch_bounds__(256) void fuse_kernel(
    const float* __restrict__ pf, const float* __restrict__ gg,
    const float* __restrict__ fu_w, const float* __restrict__ fu_g,
    const float* __restrict__ fu_b, float* __restrict__ out) {
  int o = threadIdx.x;                 // 256 channels
  const float* wr = fu_w + (size_t)o * 384;
  float f[NB];
  for (int b = 0; b < NB; ++b) {
    float s = 0.0f;
    for (int c = 0; c < 256; ++c) s += pf[b * 256 + c] * wr[c];
    for (int c = 0; c < 128; ++c) s += gg[b * 128 + c] * wr[256 + c];
    f[b] = s;
  }
  float mean = 0.25f * (f[0] + f[1] + f[2] + f[3]);
  float var = 0.0f;
  for (int b = 0; b < NB; ++b) { float d = f[b] - mean; var += d * d; }
  var *= 0.25f;
  float sc = fu_g[o] * rsqrtf(var + EPSV);
  float sh = fu_b[o] - mean * sc;
  for (int b = 0; b < NB; ++b) {
    float v = f[b] * sc + sh;
    out[b * 256 + o] = (v > 0.0f) ? v : 0.2f * v;
  }
}

// ---------------------------------------------------------------------------
extern "C" void kernel_launch(void* const* d_in, const int* in_sizes, int n_in,
                              void* d_out, int out_size, void* d_ws, size_t ws_size,
                              hipStream_t stream) {
  (void)in_sizes; (void)n_in; (void)out_size; (void)ws_size;
  const float* xyz   = (const float*)d_in[0];
  const float* pts   = (const float*)d_in[1];
  const float* ec_w  = (const float*)d_in[2];
  const float* ec_g  = (const float*)d_in[3];
  const float* ec_b  = (const float*)d_in[4];
  const float* gm_w  = (const float*)d_in[5];
  const float* gm_g  = (const float*)d_in[6];
  const float* gm_b  = (const float*)d_in[7];
  const float* w0    = (const float*)d_in[8];
  const float* gam0  = (const float*)d_in[10];
  const float* bet0  = (const float*)d_in[11];
  const float* w1    = (const float*)d_in[12];
  const float* gam1  = (const float*)d_in[14];
  const float* bet1  = (const float*)d_in[15];
  const float* w2    = (const float*)d_in[16];
  const float* gam2  = (const float*)d_in[18];
  const float* bet2  = (const float*)d_in[19];
  const float* fu_w  = (const float*)d_in[20];
  const float* fu_g  = (const float*)d_in[21];
  const float* fu_b  = (const float*)d_in[22];
  float* out = (float*)d_out;

  // ---- workspace carve-up (256B aligned) ----
  char* base = (char*)d_ws;
  size_t off = 0;
  auto alloc = [&](size_t bytes) -> void* {
    void* q = base + off;
    off += (bytes + 255) & ~(size_t)255;
    return q;
  };
  _Float16* ptsH   = (_Float16*)alloc((size_t)NB * NPTS * DF * 2);
  _Float16* g0H    = (_Float16*)alloc((size_t)BNR * 96 * 2);
  _Float16* ecwH   = (_Float16*)alloc(128 * 128 * 2);
  _Float16* gmwH   = (_Float16*)alloc(128 * 128 * 2);
  _Float16* w0H    = (_Float16*)alloc(64 * 96 * 2);
  _Float16* w1H    = (_Float16*)alloc(128 * 64 * 2);
  _Float16* w2H    = (_Float16*)alloc(256 * 128 * 2);
  int*      edges  = (int*)alloc((size_t)NB * NKR * 4);
  float* ecS = (float*)alloc(128 * 4); float* ecT = (float*)alloc(128 * 4);
  float* gmS = (float*)alloc(128 * 4); float* gmT = (float*)alloc(128 * 4);
  float* s0S = (float*)alloc(64 * 4);  float* s0T = (float*)alloc(64 * 4);
  float* s1S = (float*)alloc(128 * 4); float* s1T = (float*)alloc(128 * 4);
  float* s2S = (float*)alloc(256 * 4); float* s2T = (float*)alloc(256 * 4);
  float* pf  = (float*)alloc(NB * 256 * 4);
  float* gg  = (float*)alloc(NB * 128 * 4);
  _Float16* gfeatH = (_Float16*)alloc((size_t)BNR * 128 * 2);
  _Float16* H      = (_Float16*)alloc((size_t)NB * NKR * 128 * 2);  // 96 MB
  // H is fully consumed before gm/MLP run -> overlay later scratch on it.
  char* hb = (char*)H; size_t ho = 0;
  auto alloc2 = [&](size_t bytes) -> void* {
    void* q = hb + ho; ho += (bytes + 255) & ~(size_t)255; return q;
  };
  _Float16* gmZ = (_Float16*)alloc2((size_t)BNR * 128 * 2);
  _Float16* gmA = (_Float16*)alloc2((size_t)BNR * 128 * 2);
  _Float16* z1  = (_Float16*)alloc2((size_t)BNR * 64 * 2);
  _Float16* a1  = (_Float16*)alloc2((size_t)BNR * 64 * 2);
  _Float16* z2  = (_Float16*)alloc2((size_t)BNR * 128 * 2);
  _Float16* a2h = (_Float16*)alloc2((size_t)BNR * 128 * 2);
  _Float16* z3  = (_Float16*)alloc2((size_t)BNR * 256 * 2);
  _Float16* a3  = (_Float16*)alloc2((size_t)BNR * 256 * 2);

  auto cdiv = [](size_t a, size_t b) -> unsigned { return (unsigned)((a + b - 1) / b); };

  // ---- prep ----
  prep_pts_kernel<<<cdiv((size_t)NB * NPTS * DF, 256), 256, 0, stream>>>(pts, ptsH);
  prep_g0_kernel<<<cdiv((size_t)BNR * 96, 256), 256, 0, stream>>>(xyz, pts, g0H);
  prep_w_kernel<<<cdiv(128 * 128, 256), 256, 0, stream>>>(ec_w, ecwH, 128, 128, 128);
  prep_w_kernel<<<cdiv(128 * 128, 256), 256, 0, stream>>>(gm_w, gmwH, 128, 128, 128);
  prep_w_kernel<<<cdiv(64 * 96, 256), 256, 0, stream>>>(w0, w0H, 64, 67, 96);
  prep_w_kernel<<<cdiv(128 * 64, 256), 256, 0, stream>>>(w1, w1H, 128, 64, 64);
  prep_w_kernel<<<cdiv(256 * 128, 256), 256, 0, stream>>>(w2, w2H, 256, 128, 128);

  // ---- kNN graph ----
  knn_kernel<<<dim3(NPTS / 128, NB), 128, 0, stream>>>(xyz, edges);

  // ---- EdgeConv: h = ef @ ec_w^T  (WMMA), then BN stats + leaky + max_k ----
  edge_wmma_kernel<<<dim3(NKR / 16, 2, NB), 32, 0, stream>>>(ptsH, ecwH, edges, H);
  col_stats_kernel<<<128, 256, 0, stream>>>(H, 128, (size_t)NB * NKR, ec_g, ec_b, ecS, ecT);
  ec_apply_max_kernel<<<cdiv((size_t)BNR * 128, 256), 256, 0, stream>>>(H, ecS, ecT, gfeatH);

  // ---- graph branch: gm = gfeat @ gm_w^T, BN + leaky, max over N ----
  gemm_wmma_kernel<<<dim3(BNR / 16, 2), 32, 0, stream>>>(gfeatH, 128, gmwH, 128, gmZ, 128, 4);
  col_stats_kernel<<<128, 256, 0, stream>>>(gmZ, 128, (size_t)BNR, gm_g, gm_b, gmS, gmT);
  apply_act_kernel<<<cdiv((size_t)BNR * 128, 256), 256, 0, stream>>>(
      gmZ, gmS, gmT, gmA, (size_t)BNR * 128, 127, 0.2f);
  colmax_kernel<<<cdiv(NB * 128, 256), 256, 0, stream>>>(gmA, 128, gg);

  // ---- point MLP: 67(->96 pad) -> 64 -> 128 -> 256, BN+ReLU each ----
  gemm_wmma_kernel<<<dim3(BNR / 16, 1), 32, 0, stream>>>(g0H, 96, w0H, 96, z1, 64, 3);
  col_stats_kernel<<<64, 256, 0, stream>>>(z1, 64, (size_t)BNR, gam0, bet0, s0S, s0T);
  apply_act_kernel<<<cdiv((size_t)BNR * 64, 256), 256, 0, stream>>>(
      z1, s0S, s0T, a1, (size_t)BNR * 64, 63, 0.0f);

  gemm_wmma_kernel<<<dim3(BNR / 16, 2), 32, 0, stream>>>(a1, 64, w1H, 64, z2, 128, 2);
  col_stats_kernel<<<128, 256, 0, stream>>>(z2, 128, (size_t)BNR, gam1, bet1, s1S, s1T);
  apply_act_kernel<<<cdiv((size_t)BNR * 128, 256), 256, 0, stream>>>(
      z2, s1S, s1T, a2h, (size_t)BNR * 128, 127, 0.0f);

  gemm_wmma_kernel<<<dim3(BNR / 16, 4), 32, 0, stream>>>(a2h, 128, w2H, 128, z3, 256, 4);
  col_stats_kernel<<<256, 256, 0, stream>>>(z3, 256, (size_t)BNR, gam2, bet2, s2S, s2T);
  apply_act_kernel<<<cdiv((size_t)BNR * 256, 256), 256, 0, stream>>>(
      z3, s2S, s2T, a3, (size_t)BNR * 256, 255, 0.0f);

  colmax_kernel<<<cdiv(NB * 256, 256), 256, 0, stream>>>(a3, 256, pf);

  // ---- fusion head ----
  fuse_kernel<<<1, 256, 0, stream>>>(pf, gg, fu_w, fu_g, fu_b, out);
}
